// CompressedDoubleAttention_68736656605493
// MI455X (gfx1250) — compile-verified, matrix operand order checked by reading, not run on hardware
//
#include <hip/hip_runtime.h>

// ---------------------------------------------------------------------------
// CompressedDoubleAttention for MI455X (gfx1250, wave32, WMMA bf16 path)
// ---------------------------------------------------------------------------
constexpr int B_     = 4;
constexpr int SEQ_   = 2048;
constexpr int DIM_   = 1024;
constexpr int HEADS_ = 16;
constexpr int DH_    = 64;
constexpr int LAT_   = 512;
constexpr int HIST_  = SEQ_ - LAT_;   // 1536
constexpr int COMP_  = 512;
constexpr float SCALE_ = 0.125f;      // DH^-0.5
constexpr float EPS_   = 1e-5f;

typedef __attribute__((ext_vector_type(16))) __bf16 v16bf;
typedef __attribute__((ext_vector_type(8)))  float  v8f;
typedef __attribute__((ext_vector_type(4)))  int    v4i;

// --- gfx1250 async global->LDS copy (ASYNCcnt) if the toolchain exposes it --
#if defined(__has_builtin)
#if __has_builtin(__builtin_amdgcn_global_load_async_to_lds_b128) && \
    __has_builtin(__builtin_amdgcn_s_wait_asynccnt)
#define GFX1250_ASYNC_LDS 1
#endif
#endif
#ifndef GFX1250_ASYNC_LDS
#define GFX1250_ASYNC_LDS 0
#endif

// Copy 16B (8 x bf16) from global to LDS.
// Builtin signature (from compiler diagnostic): arg0 = AS1 v4i*, arg1 = AS3
// v4i*, then imm offset, imm cpol. Generic->AS1 is value-preserving; the low
// 32 bits of a generic LDS pointer are the LDS byte address (FLAT aperture
// rules), so the truncating cast to AS3 is well-defined.
__device__ __forceinline__ void g2l_b128(const __bf16* g, __bf16* l) {
#if GFX1250_ASYNC_LDS
  __attribute__((address_space(1))) v4i* gp =
      (__attribute__((address_space(1))) v4i*)(unsigned long long)g;
  __attribute__((address_space(3))) v4i* lp =
      (__attribute__((address_space(3))) v4i*)(unsigned int)(unsigned long long)l;
  __builtin_amdgcn_global_load_async_to_lds_b128(gp, lp, 0, 0);
#else
  *(uint4*)l = *(const uint4*)g;
#endif
}
__device__ __forceinline__ void g2l_join() {
#if GFX1250_ASYNC_LDS
  __builtin_amdgcn_s_wait_asynccnt(0);
#endif
}

// K-index of element e of a 16-bit WMMA A/B fragment (ISA 7.12.2):
// lanes 0-15 / 16-31 split the K range; elements 0-7 and 8-15 are two
// contiguous 8-element runs.
__device__ __forceinline__ int kmap(int e, int hf) {
  return (e < 8) ? (e + 8 * hf) : (e + 8 + 8 * hf);
}

// ---------------------------------------------------------------------------
// Generic tiled WMMA GEMM:  C = A * op(B) (+ bias)
//   A  : M x K row-major bf16        (+ z*sA)
//   Bm : bNT=1 -> N x K row-major (X @ W^T form)
//        bNT=0 -> K x N row-major    (+ z*sB)
//   C  : M x N, f32 (Cf) or bf16 (Cb), ld = N (+ z*sC)
//   biasMode: 0 none, 1 per-column, 2 per-row
// Macro-tile 128(M) x 64(N), K-step 32, 256 threads = 8 waves (4x2),
// each wave owns a 32x32 sub-tile = 2x2 WMMA 16x16 fragments.
// All global staging traffic is 128-bit (async-to-LDS when available).
// ---------------------------------------------------------------------------
#define BKP 40   // padded K stride in LDS (32 + 8) to dodge bank conflicts

__global__ __launch_bounds__(256)
void wmma_gemm_kernel(const __bf16* __restrict__ A,
                      const __bf16* __restrict__ Bm,
                      const float*  __restrict__ bias,
                      float*  __restrict__ Cf,
                      __bf16* __restrict__ Cb,
                      int M, int N, int K,
                      int bNT, int biasMode,
                      long long sA, long long sB, long long sC)
{
  __shared__ __bf16 As[128][BKP];
  __shared__ __bf16 Bs[64][BKP];

  const int tid  = threadIdx.x;
  const int lane = tid & 31;
  const int wv   = tid >> 5;
  const int lr   = lane & 15;
  const int hf   = lane >> 4;
  const int wm   = (wv & 3) * 32;
  const int wn   = (wv >> 2) * 32;
  const int m0   = blockIdx.y * 128;
  const int n0   = blockIdx.x * 64;
  const long long z = blockIdx.z;

  const __bf16* Az = A  + z * sA;
  const __bf16* Bz = Bm + z * sB;

  // Per-thread staging coordinates (all 16B-aligned accesses)
  const int ar = tid >> 1;            // A: row 0..127
  const int ak = (tid & 1) * 16;      // A: k-offset 0 / 16
  const int br = tid >> 2;            // B(NT): row 0..63
  const int bk = (tid & 3) * 8;       // B(NT): k-offset 0/8/16/24
  const int nk = tid & 31;            // B(NN): k index 0..31
  const int ng = (tid >> 5) * 8;      // B(NN): n group 0,8,..,56

  v8f acc[2][2];
#pragma unroll
  for (int mi = 0; mi < 2; ++mi)
#pragma unroll
    for (int ni = 0; ni < 2; ++ni)
#pragma unroll
      for (int r = 0; r < 8; ++r) acc[mi][ni][r] = 0.0f;

  for (int k0 = 0; k0 < K; k0 += 32) {
    __syncthreads();
    // ---- Stage A tile (128x32) : two b128 per thread ----
    {
      const __bf16* gA = Az + (size_t)(m0 + ar) * K + k0 + ak;
      g2l_b128(gA,     &As[ar][ak]);
      g2l_b128(gA + 8, &As[ar][ak + 8]);
      if (k0 + 32 < K) __builtin_prefetch(gA + 32, 0, 1);
    }
    // ---- Stage B tile into canonical [n][k] layout ----
    if (bNT) {
      const __bf16* gB = Bz + (size_t)(n0 + br) * K + k0 + bk;
      g2l_b128(gB, &Bs[br][bk]);
      if (k0 + 32 < K) __builtin_prefetch(gB + 32, 0, 1);
    } else {
      // K-major source: one b128 of 8 consecutive n, transpose-scatter to LDS
      const __bf16* gB = Bz + (size_t)(k0 + nk) * N + n0 + ng;
      union { uint4 u; __bf16 h[8]; } t;
      t.u = *(const uint4*)gB;
#pragma unroll
      for (int j = 0; j < 8; ++j) Bs[ng + j][nk] = t.h[j];
      if (k0 + 32 < K)
        __builtin_prefetch(Bz + (size_t)(k0 + 32 + nk) * N + n0 + ng, 0, 1);
    }
    g2l_join();
    __syncthreads();

    v16bf af[2], bfg[2];
#pragma unroll
    for (int mi = 0; mi < 2; ++mi) {
      const __bf16* p = &As[wm + mi * 16 + lr][0];
#pragma unroll
      for (int e = 0; e < 16; ++e) af[mi][e] = p[kmap(e, hf)];
    }
#pragma unroll
    for (int ni = 0; ni < 2; ++ni) {
      const __bf16* p = &Bs[wn + ni * 16 + lr][0];
#pragma unroll
      for (int e = 0; e < 16; ++e) bfg[ni][e] = p[kmap(e, hf)];
    }
#pragma unroll
    for (int mi = 0; mi < 2; ++mi)
#pragma unroll
      for (int ni = 0; ni < 2; ++ni)
        acc[mi][ni] = __builtin_amdgcn_wmma_f32_16x16x32_bf16(
            false, af[mi], false, bfg[ni], (short)0, acc[mi][ni], false, false);
  }

  // Epilogue
#pragma unroll
  for (int mi = 0; mi < 2; ++mi)
#pragma unroll
    for (int ni = 0; ni < 2; ++ni)
#pragma unroll
      for (int r = 0; r < 8; ++r) {
        const int row = m0 + wm + mi * 16 + r + 8 * hf;
        const int col = n0 + wn + ni * 16 + lr;
        float v = acc[mi][ni][r];
        if (biasMode == 1)      v += bias[col];
        else if (biasMode == 2) v += bias[row];
        const size_t off = (size_t)z * sC + (size_t)row * N + col;
        if (Cf) Cf[off] = v;
        else    Cb[off] = (__bf16)v;
      }
}

// ---------------------------------------------------------------------------
// Flash attention: per-wave 16-row q tile, 32-wide kv tiles staged in LDS,
// online softmax, PV accumulation via WMMA. q/kv layouts: (B,H,SEQ,DH) bf16.
// out: (B, q_len, DIM) f32 at local row index.
// ---------------------------------------------------------------------------
__global__ __launch_bounds__(256)
void flash_attn_kernel(const __bf16* __restrict__ qb,
                       const __bf16* __restrict__ kvb,
                       float* __restrict__ outp,
                       int q_start, int q_len, int kv_len, int causal)
{
  __shared__ __bf16 kv_lds[8][32][64];     // per-wave kv tile (4KB each)
  __shared__ float  p_lds[8][16][36];      // per-wave P bounce buffer

  const int tid  = threadIdx.x;
  const int wv   = tid >> 5;
  const int lane = tid & 31;
  const int lr   = lane & 15;
  const int hf   = lane >> 4;
  const int bh   = blockIdx.y;
  const int b    = bh / HEADS_;
  const int h    = bh % HEADS_;
  const int tile0 = (blockIdx.x * 8 + wv) * 16;   // local q row base

  const size_t headBase = ((size_t)(b * HEADS_ + h)) * SEQ_ * DH_;

  // Preload q A-fragments (16x64 -> two K=32 fragments)
  v16bf aq[2];
  {
    const __bf16* qp = qb + headBase + (size_t)(q_start + tile0 + lr) * DH_;
#pragma unroll
    for (int ks = 0; ks < 2; ++ks)
#pragma unroll
      for (int e = 0; e < 16; ++e)
        aq[ks][e] = qp[32 * ks + kmap(e, hf)];
  }

  float mrow[8], lsum[8];
  v8f acc[4];
#pragma unroll
  for (int r = 0; r < 8; ++r) { mrow[r] = -3.0e38f; lsum[r] = 0.0f; }
#pragma unroll
  for (int t = 0; t < 4; ++t)
#pragma unroll
    for (int r = 0; r < 8; ++r) acc[t][r] = 0.0f;

  const int qmax = q_start + tile0 + 15;
  for (int j0 = 0; j0 < kv_len; j0 += 32) {
    if (causal && j0 > qmax) break;               // wave-uniform early exit

    // Stage kv tile: lane L copies row j0+L (128B) into LDS, all b128
    {
      const __bf16* gk = kvb + headBase + (size_t)(j0 + lane) * DH_;
      __bf16* lk = &kv_lds[wv][lane][0];
#pragma unroll
      for (int c = 0; c < 8; ++c) g2l_b128(gk + 8 * c, lk + 8 * c);
      if (j0 + 32 < kv_len) __builtin_prefetch(gk + 32 * DH_, 0, 1);
      g2l_join();
    }

    // S = q(16x64) @ kv^T : two 16-column sub-tiles, K accumulated over 64
    v8f S[2];
#pragma unroll
    for (int sub = 0; sub < 2; ++sub) {
      v16bf bk2[2];
      const __bf16* kp = &kv_lds[wv][sub * 16 + lr][0];
#pragma unroll
      for (int ks = 0; ks < 2; ++ks)
#pragma unroll
        for (int e = 0; e < 16; ++e)
          bk2[ks][e] = kp[32 * ks + kmap(e, hf)];
      v8f s;
#pragma unroll
      for (int r = 0; r < 8; ++r) s[r] = 0.0f;
      s = __builtin_amdgcn_wmma_f32_16x16x32_bf16(false, aq[0], false, bk2[0],
                                                  (short)0, s, false, false);
      s = __builtin_amdgcn_wmma_f32_16x16x32_bf16(false, aq[1], false, bk2[1],
                                                  (short)0, s, false, false);
      S[sub] = s;
    }

    // Mask + online softmax. C layout: col = lane%16, row = r + 8*(lane/16);
    // row reductions via shfl-xor 1..8 (stay inside 16-lane halves).
#pragma unroll
    for (int r = 0; r < 8; ++r) {
      const int rowg = q_start + tile0 + r + 8 * hf;
      float v0 = S[0][r], v1 = S[1][r];
      const int c0 = j0 + lr, c1 = j0 + 16 + lr;
      if (c0 >= kv_len || (causal && c0 > rowg)) v0 = -3.0e38f;
      if (c1 >= kv_len || (causal && c1 > rowg)) v1 = -3.0e38f;
      float mx = fmaxf(v0, v1);
#pragma unroll
      for (int o = 1; o < 16; o <<= 1) mx = fmaxf(mx, __shfl_xor(mx, o, 32));
      const float mnew  = fmaxf(mrow[r], mx);
      const float alpha = __expf(mrow[r] - mnew);
      mrow[r] = mnew;
      const float p0 = __expf(v0 - mnew);
      const float p1 = __expf(v1 - mnew);
      float sm = p0 + p1;
#pragma unroll
      for (int o = 1; o < 16; o <<= 1) sm += __shfl_xor(sm, o, 32);
      lsum[r] = lsum[r] * alpha + sm;
#pragma unroll
      for (int t = 0; t < 4; ++t) acc[t][r] *= alpha;
      p_lds[wv][r + 8 * hf][lr]      = p0;
      p_lds[wv][r + 8 * hf][16 + lr] = p1;
    }

    // PV: P(16x32) @ kv(32x64); P re-read in A layout through LDS bounce.
    v16bf pa;
#pragma unroll
    for (int e = 0; e < 16; ++e)
      pa[e] = (__bf16)p_lds[wv][lr][kmap(e, hf)];
#pragma unroll
    for (int t = 0; t < 4; ++t) {
      v16bf bv;
#pragma unroll
      for (int e = 0; e < 16; ++e)
        bv[e] = kv_lds[wv][kmap(e, hf)][t * 16 + lr];
      acc[t] = __builtin_amdgcn_wmma_f32_16x16x32_bf16(
          false, pa, false, bv, (short)0, acc[t], false, false);
    }
  }

  // Normalize and scatter to (B, q_len, DIM) f32
#pragma unroll
  for (int r = 0; r < 8; ++r) {
    const float inv = 1.0f / lsum[r];
    const int row = tile0 + r + 8 * hf;
    float* op = outp + ((size_t)b * q_len + row) * DIM_ + h * DH_;
#pragma unroll
    for (int t = 0; t < 4; ++t) op[t * 16 + lr] = acc[t][r] * inv;
  }
}

// ---------------------------------------------------------------------------
// Elementwise kernels
// ---------------------------------------------------------------------------
__global__ void f32_to_bf16_kernel(const float* __restrict__ src,
                                   __bf16* __restrict__ dst, long long n)
{
  const long long i = (long long)blockIdx.x * blockDim.x + threadIdx.x;
  if (i < n) dst[i] = (__bf16)src[i];
}

// out_lat (B,LAT,DIM) f32 -> concat buffer (B, COMP+LAT, DIM) bf16 rows [COMP..)
__global__ void lat_to_concat_kernel(const float* __restrict__ src,
                                     __bf16* __restrict__ dst)
{
  const long long n = (long long)B_ * LAT_ * DIM_;
  const long long i = (long long)blockIdx.x * blockDim.x + threadIdx.x;
  if (i >= n) return;
  const int d = (int)(i % DIM_);
  const long long t = i / DIM_;
  const int r = (int)(t % LAT_);
  const int b = (int)(t / LAT_);
  dst[((size_t)b * (COMP_ + LAT_) + COMP_ + r) * DIM_ + d] = (__bf16)src[i];
}

// q_lin (B*SEQ, DIM) f32 -> rotary + scale -> q_bf (B,H,SEQ,DH) bf16
__global__ void rot_q_kernel(const float* __restrict__ qlin,
                             __bf16* __restrict__ qb)
{
  const int idx = blockIdx.x * blockDim.x + threadIdx.x;
  if (idx >= B_ * SEQ_ * HEADS_) return;
  const int h  = idx % HEADS_;
  const int bs = idx / HEADS_;
  const int s  = bs % SEQ_;
  const int b  = bs / SEQ_;
  const float* src = qlin + (size_t)bs * DIM_ + h * DH_;
  __bf16* dst = qb + ((size_t)(b * HEADS_ + h) * SEQ_ + s) * DH_;
  float t[DH_];
#pragma unroll
  for (int d = 0; d < DH_; ++d) t[d] = src[d];
#pragma unroll
  for (int i = 0; i < DH_ / 2; ++i) {
    const float ang = (float)s * __powf(10000.0f, -((float)(2 * i)) / (float)DH_);
    const float c = __cosf(ang), sn = __sinf(ang);
    const float lo = t[i], hi = t[i + DH_ / 2];
    dst[i]           = (__bf16)((lo * c - hi * sn) * SCALE_);
    dst[i + DH_ / 2] = (__bf16)((hi * c + lo * sn) * SCALE_);
  }
}

// kv_lin (B*SEQ, DIM) f32 -> rotary + layernorm(DH) -> kv_bf (B,H,SEQ,DH) bf16
__global__ void rot_ln_kv_kernel(const float* __restrict__ kvlin,
                                 const float* __restrict__ gamma,
                                 const float* __restrict__ beta,
                                 __bf16* __restrict__ kvb)
{
  const int idx = blockIdx.x * blockDim.x + threadIdx.x;
  if (idx >= B_ * SEQ_ * HEADS_) return;
  const int h  = idx % HEADS_;
  const int bs = idx / HEADS_;
  const int s  = bs % SEQ_;
  const int b  = bs / SEQ_;
  const float* src = kvlin + (size_t)bs * DIM_ + h * DH_;
  __bf16* dst = kvb + ((size_t)(b * HEADS_ + h) * SEQ_ + s) * DH_;
  float o[DH_];
#pragma unroll
  for (int i = 0; i < DH_ / 2; ++i) {
    const float ang = (float)s * __powf(10000.0f, -((float)(2 * i)) / (float)DH_);
    const float c = __cosf(ang), sn = __sinf(ang);
    const float lo = src[i], hi = src[i + DH_ / 2];
    o[i]           = lo * c - hi * sn;
    o[i + DH_ / 2] = hi * c + lo * sn;
  }
  float mu = 0.0f;
#pragma unroll
  for (int d = 0; d < DH_; ++d) mu += o[d];
  mu *= (1.0f / DH_);
  float var = 0.0f;
#pragma unroll
  for (int d = 0; d < DH_; ++d) { const float dd = o[d] - mu; var += dd * dd; }
  var *= (1.0f / DH_);
  const float rinv = rsqrtf(var + EPS_);
#pragma unroll
  for (int d = 0; d < DH_; ++d)
    dst[d] = (__bf16)((o[d] - mu) * rinv * gamma[d] + beta[d]);
}

// ---------------------------------------------------------------------------
// Host launcher
// ---------------------------------------------------------------------------
extern "C" void kernel_launch(void* const* d_in, const int* in_sizes, int n_in,
                              void* d_out, int out_size, void* d_ws, size_t ws_size,
                              hipStream_t stream)
{
  (void)in_sizes; (void)n_in; (void)out_size; (void)ws_size;

  const float* x      = (const float*)d_in[0];
  const float* Wq     = (const float*)d_in[1];
  const float* Wkv    = (const float*)d_in[2];
  const float* Wout   = (const float*)d_in[3];
  const float* b_out  = (const float*)d_in[4];
  const float* Wproj  = (const float*)d_in[5];
  const float* b_proj = (const float*)d_in[6];
  const float* gamma  = (const float*)d_in[7];
  const float* beta   = (const float*)d_in[8];
  float* out = (float*)d_out;

  char* ws = (char*)d_ws;
  size_t cur = 0;
  auto wsAlloc = [&](size_t bytes) -> void* {
    void* p = ws + cur;
    cur += (bytes + 255) & ~(size_t)255;
    return p;
  };

  const size_t nX = (size_t)B_ * SEQ_ * DIM_;              // 8.39M
  __bf16* xb        = (__bf16*)wsAlloc(nX * 2);
  __bf16* Wq_b      = (__bf16*)wsAlloc((size_t)DIM_ * DIM_ * 2);
  __bf16* Wkv_b     = (__bf16*)wsAlloc((size_t)DIM_ * DIM_ * 2);
  __bf16* Wout_b    = (__bf16*)wsAlloc((size_t)DIM_ * DIM_ * 2);
  __bf16* Wproj_b   = (__bf16*)wsAlloc((size_t)COMP_ * HIST_ * 2);
  float*  q_lin     = (float*)wsAlloc(nX * 4);
  float*  kv_lin    = (float*)wsAlloc(nX * 4);
  __bf16* q_bf      = (__bf16*)wsAlloc(nX * 2);
  __bf16* kv_bf     = (__bf16*)wsAlloc(nX * 2);
  float*  out_hist  = (float*)wsAlloc((size_t)B_ * HIST_ * DIM_ * 4);
  float*  out_lat   = (float*)wsAlloc((size_t)B_ * LAT_ * DIM_ * 4);
  __bf16* out_hist_b= (__bf16*)wsAlloc((size_t)B_ * HIST_ * DIM_ * 2);
  __bf16* outc_b    = (__bf16*)wsAlloc((size_t)B_ * (COMP_ + LAT_) * DIM_ * 2);

  auto cvt = [&](const float* s, __bf16* d, long long n) {
    const int blk = 256;
    const int grd = (int)((n + blk - 1) / blk);
    f32_to_bf16_kernel<<<grd, blk, 0, stream>>>(s, d, n);
  };

  // 1) bf16 conversions of inputs/weights
  cvt(x,     xb,      (long long)nX);
  cvt(Wq,    Wq_b,    (long long)DIM_ * DIM_);
  cvt(Wkv,   Wkv_b,   (long long)DIM_ * DIM_);
  cvt(Wout,  Wout_b,  (long long)DIM_ * DIM_);
  cvt(Wproj, Wproj_b, (long long)COMP_ * HIST_);

  // 2) q/kv projections: (B*SEQ,1024) = xb @ W^T
  {
    dim3 grd(DIM_ / 64, (B_ * SEQ_) / 128, 1);
    wmma_gemm_kernel<<<grd, 256, 0, stream>>>(xb, Wq_b, nullptr, q_lin, nullptr,
        B_ * SEQ_, DIM_, DIM_, /*bNT=*/1, /*bias=*/0, 0, 0, 0);
    wmma_gemm_kernel<<<grd, 256, 0, stream>>>(xb, Wkv_b, nullptr, kv_lin, nullptr,
        B_ * SEQ_, DIM_, DIM_, 1, 0, 0, 0, 0);
  }

  // 3) rotary (+scale) on q, rotary + layernorm on kv, to (B,H,SEQ,DH) bf16
  {
    const int total = B_ * SEQ_ * HEADS_;
    rot_q_kernel<<<(total + 255) / 256, 256, 0, stream>>>(q_lin, q_bf);
    rot_ln_kv_kernel<<<(total + 255) / 256, 256, 0, stream>>>(kv_lin, gamma, beta, kv_bf);
  }

  // 4) attention
  {
    dim3 grdH(HIST_ / 128, B_ * HEADS_, 1);   // hist: q[0,HIST) x kv[0,HIST), no mask
    flash_attn_kernel<<<grdH, 256, 0, stream>>>(q_bf, kv_bf, out_hist,
        /*q_start=*/0, /*q_len=*/HIST_, /*kv_len=*/HIST_, /*causal=*/0);
    dim3 grdL(LAT_ / 128, B_ * HEADS_, 1);    // latent: q[HIST,SEQ) x kv[0,SEQ), causal
    flash_attn_kernel<<<grdL, 256, 0, stream>>>(q_bf, kv_bf, out_lat,
        HIST_, LAT_, SEQ_, 1);
  }

  // 5) sequence compression: per batch, C(COMP,DIM) = Wproj @ out_hist + b_proj
  cvt(out_hist, out_hist_b, (long long)B_ * HIST_ * DIM_);
  {
    dim3 grd(DIM_ / 64, COMP_ / 128, B_);
    wmma_gemm_kernel<<<grd, 256, 0, stream>>>(Wproj_b, out_hist_b, b_proj,
        nullptr, outc_b,
        COMP_, DIM_, HIST_, /*bNT=*/0, /*biasMode=*/2,
        /*sA=*/0, /*sB=*/(long long)HIST_ * DIM_,
        /*sC=*/(long long)(COMP_ + LAT_) * DIM_);
  }

  // 6) append latent rows into concat buffer
  {
    const long long n = (long long)B_ * LAT_ * DIM_;
    lat_to_concat_kernel<<<(int)((n + 255) / 256), 256, 0, stream>>>(out_lat, outc_b);
  }

  // 7) final projection: (B*1024, DIM) = outc @ Wout^T + b_out -> d_out f32
  {
    dim3 grd(DIM_ / 64, (B_ * (COMP_ + LAT_)) / 128, 1);
    wmma_gemm_kernel<<<grd, 256, 0, stream>>>(outc_b, Wout_b, b_out, out, nullptr,
        B_ * (COMP_ + LAT_), DIM_, DIM_, /*bNT=*/1, /*biasMode=*/1, 0, 0, 0);
  }
}
